// ReadoutFunction_8796093022561
// MI455X (gfx1250) — compile-verified, matrix operand order checked by reading, not run on hardware
//
#include <hip/hip_runtime.h>

// ---------------------------------------------------------------------------
// Problem constants (from reference)
// ---------------------------------------------------------------------------
#define N_NODE  200000
#define N_DIM   128
#define HID     512
#define G_DIM   128
#define N_GRAPH 1024
#define TILES   (N_NODE / 16)   // 12500, exact

typedef __attribute__((ext_vector_type(16))) __bf16 v16bf;
typedef __attribute__((ext_vector_type(8)))  __bf16 v8bf;
typedef __attribute__((ext_vector_type(8)))  float  v8f;
typedef __attribute__((ext_vector_type(4)))  unsigned int v4u;
typedef __attribute__((ext_vector_type(8)))  int    v8i;
typedef __attribute__((ext_vector_type(4)))  int    v4i;

#if __has_builtin(__builtin_amdgcn_tensor_load_to_lds) && \
    __has_builtin(__builtin_amdgcn_s_wait_tensorcnt)
#define USE_TDM 1
#else
#define USE_TDM 0
#endif

// round-to-nearest-even f32 -> bf16
__device__ __forceinline__ __bf16 f2bf(float f) {
  unsigned u = __builtin_bit_cast(unsigned, f);
  u += 0x7fffu + ((u >> 16) & 1u);
  unsigned short h = (unsigned short)(u >> 16);
  return __builtin_bit_cast(__bf16, h);
}

__device__ __forceinline__ v8f splat8(float s) {
  v8f r;
#pragma unroll
  for (int i = 0; i < 8; ++i) r[i] = s;
  return r;
}

__device__ __forceinline__ v16bf load_frag(const __bf16* p) {
  return *(const v16bf*)p;   // 32B per lane, contiguous across the wave
}

#if USE_TDM
// ---------------------------------------------------------------------------
// Issue one TDM 2D tile load (rows x cols f32, row-major) into LDS.
// D# per CDNA5 ISA §8: group0 = {count=1 | lds_addr | global_addr | type=2},
// group1 = {data_size=4B, tensor_dim0/1, tile_dim0/1, tensor_dim0_stride}.
// Groups 2/3 are only needed for >2D tensors -> zero-filled here.
// This toolchain exposes the 6-arg builtin:
//   (uint32x4 g0, int32x8 g1, int32x4, int32x4, int32x8, i32 cpol)
// EXEC is ignored for tensor ops; descriptor values are wave-uniform.
// ---------------------------------------------------------------------------
__device__ __forceinline__ void tdm_load_tile_2d(const float* gptr,
                                                 unsigned lds_byte_off,
                                                 int rows, int cols,
                                                 int row_stride_elems) {
  unsigned long long ga = (unsigned long long)(size_t)gptr;
  v4u g0;
  g0[0] = 1u;                                   // count=1 (valid user D#)
  g0[1] = lds_byte_off;                         // lds_addr (bytes)
  g0[2] = (unsigned)(ga & 0xffffffffu);         // global_addr[31:0]
  g0[3] = (unsigned)((ga >> 32) & 0x01ffffffu)  // global_addr[56:32]
          | (2u << 30);                         // type=2 ("image")
  v8i g1;
  g1[0] = (int)(2u << 16);                                  // data_size=4B
  g1[1] = (int)(((unsigned)cols & 0xffffu) << 16);          // tensor_dim0 lo16
  g1[2] = (int)((((unsigned)cols >> 16) & 0xffffu) |
                (((unsigned)rows & 0xffffu) << 16));        // dim0 hi | dim1 lo
  g1[3] = (int)((((unsigned)rows >> 16) & 0xffffu) |
                (((unsigned)cols & 0xffffu) << 16));        // dim1 hi | tile_dim0
  g1[4] = rows;                                             // tile_dim1 (tile_dim2=0)
  g1[5] = row_stride_elems;                                 // tensor_dim0_stride lo
  g1[6] = 0;
  g1[7] = 0;
  v4i z4 = {0, 0, 0, 0};
  v8i z8 = {0, 0, 0, 0, 0, 0, 0, 0};
  __builtin_amdgcn_tensor_load_to_lds(g0, g1, z4, z4, z8, 0);
}
#endif

// ---------------------------------------------------------------------------
// Zero the [N_GRAPH, G_DIM] output accumulator (harness poisons d_out).
// ---------------------------------------------------------------------------
__global__ void zero_out_kernel(float* __restrict__ out, int n) {
  int i = blockIdx.x * blockDim.x + threadIdx.x;
  if (i < n) out[i] = 0.f;
}

// ---------------------------------------------------------------------------
// Re-swizzle a row-major f32 weight matrix W[K][N] into bf16 WMMA B-fragments.
// One fragment is a 32x16 (KxN) tile stored as 512 contiguous bf16:
// element index = lane*16+e, lane = n + 16*khalf,
// klocal = (v<4 ? 2v : 16+2(v-4)) + 8*khalf + (e&1).  Order: [kchunk][ntile].
// ---------------------------------------------------------------------------
__global__ void prep_wfrag_kernel(const float* __restrict__ W,
                                  __bf16* __restrict__ dst, int K, int N) {
  int idx = blockIdx.x * blockDim.x + threadIdx.x;
  if (idx >= K * N) return;
  int frag   = idx >> 9;        // /512
  int r      = idx & 511;
  int lane   = r >> 4;
  int e      = r & 15;
  int ntiles = N >> 4;
  int kchunk = frag / ntiles;
  int ntile  = frag % ntiles;
  int n      = lane & 15;
  int khalf  = lane >> 4;
  int v      = e >> 1;
  int klocal = (v < 4 ? 2 * v : 16 + 2 * (v - 4)) + 8 * khalf + (e & 1);
  int k      = kchunk * 32 + klocal;
  int col    = ntile * 16 + n;
  dst[idx]   = f2bf(W[(size_t)k * N + col]);
}

// ---------------------------------------------------------------------------
// Fused MLP pass: x[16,32*NK] @ W1 -> ReLU -> @ W2, accumulated into acc[8]
// (16x128 f32 C-layout). Hidden (512) processed in 32-column chunks; the
// C-layout -> A-layout transpose between layers goes through a wave-private
// 16x32 bf16 LDS tile (LDS is in-order per wave; no barrier needed).
// ---------------------------------------------------------------------------
template <int NK>
__device__ __forceinline__ void mlp_pass(const v16bf a[8],
                                         const __bf16* __restrict__ w1frag,
                                         const float*  __restrict__ b1,
                                         const __bf16* __restrict__ w2frag,
                                         v8f acc[8], __bf16* lb, int lane) {
  const int ncol  = lane & 15;
  const int khalf = lane >> 4;
  const int m     = lane & 15;
#pragma unroll 1
  for (int hc = 0; hc < 16; ++hc) {
    v8f h0a = splat8(b1[hc * 32 + ncol]);
    v8f h1a = splat8(b1[hc * 32 + 16 + ncol]);
#pragma unroll
    for (int k = 0; k < NK; ++k) {
      const __bf16* base = w1frag + ((size_t)(k * 32 + 2 * hc) << 9) + lane * 16;
      v16bf bf0 = load_frag(base);
      v16bf bf1 = load_frag(base + 512);
      h0a = __builtin_amdgcn_wmma_f32_16x16x32_bf16(false, a[k], false, bf0,
                                                    (short)0, h0a, false, false);
      h1a = __builtin_amdgcn_wmma_f32_16x16x32_bf16(false, a[k], false, bf1,
                                                    (short)0, h1a, false, false);
    }
    // ReLU + transpose C-layout -> A-layout via LDS (bf16 16x32 tile)
#pragma unroll
    for (int v = 0; v < 8; ++v) {
      int mr = v + 8 * khalf;
      lb[mr * 32 + ncol]      = f2bf(fmaxf(h0a[v], 0.f));
      lb[mr * 32 + 16 + ncol] = f2bf(fmaxf(h1a[v], 0.f));
    }
    v8bf lo = *(const v8bf*)(lb + m * 32 + 8 * khalf);        // ds_load_b128
    v8bf hi = *(const v8bf*)(lb + m * 32 + 16 + 8 * khalf);   // ds_load_b128
    v16bf hf = __builtin_shufflevector(lo, hi, 0, 1, 2, 3, 4, 5, 6, 7,
                                       8, 9, 10, 11, 12, 13, 14, 15);
#pragma unroll
    for (int t = 0; t < 8; ++t) {
      v16bf bw = load_frag(w2frag + ((size_t)(hc * 8 + t) << 9) + lane * 16);
      acc[t] = __builtin_amdgcn_wmma_f32_16x16x32_bf16(false, hf, false, bw,
                                                       (short)0, acc[t], false, false);
    }
  }
}

// ---------------------------------------------------------------------------
// Main fused kernel: one wave per 16-node tile, 4 waves per block.
// Activations staged to LDS by the Tensor Data Mover when available.
// ---------------------------------------------------------------------------
__global__ __launch_bounds__(128, 1) void fused_readout_kernel(
    const float* __restrict__ hT, const float* __restrict__ h0,
    const int*   __restrict__ gidx,
    const __bf16* __restrict__ wi1, const float* __restrict__ bi1,
    const __bf16* __restrict__ wi2, const float* __restrict__ bi2,
    const __bf16* __restrict__ wj1, const float* __restrict__ bj1,
    const __bf16* __restrict__ wj2, const float* __restrict__ bj2,
    float* __restrict__ out) {
  __shared__ __bf16 ldsT[4][16 * 32];          // transpose staging, 1KB/wave
#if USE_TDM
  __shared__ float sx[4][2][16 * 128];         // activation tiles, 16KB/wave
#endif
  const int lane  = threadIdx.x & 31;
  const int wave  = threadIdx.x >> 5;
  const int tile  = blockIdx.x * 4 + wave;
  if (tile >= TILES) return;
  const int r0    = tile * 16;
  const int m     = lane & 15;
  const int khalf = lane >> 4;
  const int ncol  = lane & 15;
  __bf16* lb = &ldsT[wave][0];

#if USE_TDM
  // DMA both 16x128 f32 activation tiles into this wave's LDS region.
  {
    unsigned base = (unsigned)(size_t)(&sx[wave][0][0]);   // LDS byte offset
    tdm_load_tile_2d(hT + (size_t)r0 * N_DIM, base, 16, N_DIM, N_DIM);
    tdm_load_tile_2d(h0 + (size_t)r0 * N_DIM, base + 16 * N_DIM * 4,
                     16, N_DIM, N_DIM);
  }
#else
  // speculative prefetch of next tile's activations (global_prefetch_b8)
  if (tile + 1 < TILES) {
    __builtin_prefetch(hT + (size_t)(r0 + 16 + m) * N_DIM, 0, 1);
    __builtin_prefetch(h0 + (size_t)(r0 + 16 + m) * N_DIM, 0, 1);
  }
#endif

  // Prefetch graph indices while the DMA is in flight.
  int gv[8];
#pragma unroll
  for (int v = 0; v < 8; ++v) gv[v] = gidx[r0 + khalf * 8 + v];

#if USE_TDM
  __builtin_amdgcn_s_wait_tensorcnt(0);        // s_wait_tensorcnt 0
#endif

  // Build x = concat(h_T, h_0)[16 rows] as 8 bf16 A-fragments (K=32 each).
  v16bf a[8];
#pragma unroll
  for (int c = 0; c < 8; ++c) {
#if USE_TDM
    const float* src = &sx[wave][c < 4 ? 0 : 1][m * N_DIM + (c & 3) * 32];
#else
    const float* src =
        (c < 4 ? hT : h0) + (size_t)(r0 + m) * N_DIM + (c & 3) * 32;
#endif
    v8f lo = *(const v8f*)(src + 8 * khalf);
    v8f hi = *(const v8f*)(src + 16 + 8 * khalf);
    v16bf av;
#pragma unroll
    for (int e = 0; e < 8; ++e) { av[e] = f2bf(lo[e]); av[8 + e] = f2bf(hi[e]); }
    a[c] = av;
  }

  // C/D accumulators [16 x 128] for both MLPs, bias-initialized.
  v8f iacc[8], jacc[8];
#pragma unroll
  for (int t = 0; t < 8; ++t) {
    iacc[t] = splat8(bi2[t * 16 + ncol]);
    jacc[t] = splat8(bj2[t * 16 + ncol]);
  }
  mlp_pass<8>(a, wi1, bi1, wi2, iacc, lb, lane);  // i_net: K = 256
  mlp_pass<4>(a, wj1, bj1, wj2, jacc, lb, lane);  // j_net: K = 128 (h_T only)

  // Segment-sum: lane holds 8 consecutive node rows (m = v + 8*khalf) per
  // column; graph_index is sorted -> run-length reduce, then atomic add.
#pragma unroll
  for (int t = 0; t < 8; ++t) {
    float run = 0.f;
    int gcur = gv[0];
#pragma unroll
    for (int v = 0; v < 8; ++v) {
      float pre  = iacc[t][v];
      float gate = __builtin_amdgcn_rcpf(
          1.0f + __builtin_amdgcn_exp2f(-pre * 1.4426950408889634f));
      float val = gate * jacc[t][v];
      if (gv[v] != gcur) {
        atomicAdd(out + (size_t)gcur * G_DIM + t * 16 + ncol, run);
        run = 0.f;
        gcur = gv[v];
      }
      run += val;
    }
    atomicAdd(out + (size_t)gcur * G_DIM + t * 16 + ncol, run);
  }
}

// ---------------------------------------------------------------------------
// Launcher. d_in order: h_T, h_0, graph_index, Wi1, bi1, Wi2, bi2,
//                       Wj1, bj1, Wj2, bj2
// ---------------------------------------------------------------------------
extern "C" void kernel_launch(void* const* d_in, const int* in_sizes, int n_in,
                              void* d_out, int out_size, void* d_ws,
                              size_t ws_size, hipStream_t stream) {
  const float* hT  = (const float*)d_in[0];
  const float* h0  = (const float*)d_in[1];
  const int*   gix = (const int*)d_in[2];
  const float* Wi1 = (const float*)d_in[3];
  const float* bi1 = (const float*)d_in[4];
  const float* Wi2 = (const float*)d_in[5];
  const float* bi2 = (const float*)d_in[6];
  const float* Wj1 = (const float*)d_in[7];
  const float* bj1 = (const float*)d_in[8];
  const float* Wj2 = (const float*)d_in[9];
  const float* bj2 = (const float*)d_in[10];

  // Workspace layout (bf16 elements): swizzled weight fragments.
  __bf16* wb    = (__bf16*)d_ws;
  __bf16* wb_i1 = wb;                 // 2*N_DIM x HID   = 131072
  __bf16* wb_i2 = wb + 131072;        // HID x G_DIM     =  65536
  __bf16* wb_j1 = wb + 196608;        // N_DIM x HID     =  65536
  __bf16* wb_j2 = wb + 262144;        // HID x G_DIM     =  65536

  zero_out_kernel<<<(N_GRAPH * G_DIM + 255) / 256, 256, 0, stream>>>(
      (float*)d_out, N_GRAPH * G_DIM);

  prep_wfrag_kernel<<<(2 * N_DIM * HID + 255) / 256, 256, 0, stream>>>(
      Wi1, wb_i1, 2 * N_DIM, HID);
  prep_wfrag_kernel<<<(HID * G_DIM + 255) / 256, 256, 0, stream>>>(
      Wi2, wb_i2, HID, G_DIM);
  prep_wfrag_kernel<<<(N_DIM * HID + 255) / 256, 256, 0, stream>>>(
      Wj1, wb_j1, N_DIM, HID);
  prep_wfrag_kernel<<<(HID * G_DIM + 255) / 256, 256, 0, stream>>>(
      Wj2, wb_j2, HID, G_DIM);

  fused_readout_kernel<<<TILES / 4, 128, 0, stream>>>(
      hT, h0, gix, wb_i1, bi1, wb_i2, bi2, wb_j1, bj1, wb_j2, bj2,
      (float*)d_out);
}
